// AGCRNCell_74328704024842
// MI455X (gfx1250) — matrix operand order repaired and checked by reading, not compiled, not run on hardware
//
#include <hip/hip_runtime.h>
#include <hip/hip_bf16.h>
#include <math.h>

// ---------------------------------------------------------------------------
// AGCRN cell for MI455X (gfx1250, wave32, WMMA).
// B=64, N=2048, DIN=2, DOUT=64, D=10, CHEB_K=2.
//
// Heavy math on v_wmma_f32_16x16x32_f16 (f16 in / f32 accum):
//   - AGG = softmax(relu(E E^T)) @ node-major activations  (one big GEMM,
//     double-buffered LDS, A tiles staged with global_load_async_to_lds_b128)
//   - per-node [64 x 160] @ [160 x {128,64}] with per-node weights
//     W_n = sum_d E[n,d] * w_pool[d] synthesized into LDS.
// ---------------------------------------------------------------------------

typedef __attribute__((ext_vector_type(16))) _Float16 v16h;
typedef __attribute__((ext_vector_type(8)))  float    v8f;

#define B_      64
#define N_      2048
#define DIN_    2
#define DOUT_   64
#define DD_     10
#define CIN_    66            // DIN + DOUT
#define COLS_   (B_ * CIN_)   // 4224, multiple of 128
#define FTOT_   132           // CHEB_K * CIN
#define FP_     160           // FTOT padded to multiple of 32 (WMMA K step)
#define COUTG_  128           // gate out (2*DOUT)
#define COUTU_  64            // update out

// ---------------------------------------------------------------------------
// Kernel 1: S[n][m] = softmax_m( relu(E[n] . E[m]) ), stored f16 row-major.
// ---------------------------------------------------------------------------
__global__ __launch_bounds__(256)
void supports_softmax_kernel(const float* __restrict__ E, _Float16* __restrict__ S)
{
    __shared__ float row[N_];
    __shared__ float red[256];
    const int n   = blockIdx.x;
    const int tid = threadIdx.x;
    float e[DD_];
#pragma unroll
    for (int d = 0; d < DD_; ++d) e[d] = E[n * DD_ + d];

    float lmax = 0.0f;
    for (int m = tid; m < N_; m += 256) {
        float acc = 0.0f;
#pragma unroll
        for (int d = 0; d < DD_; ++d) acc += e[d] * E[m * DD_ + d];
        acc = fmaxf(acc, 0.0f);              // relu
        row[m] = acc;
        lmax = fmaxf(lmax, acc);
    }
    red[tid] = lmax; __syncthreads();
    for (int s = 128; s > 0; s >>= 1) {
        if (tid < s) red[tid] = fmaxf(red[tid], red[tid + s]);
        __syncthreads();
    }
    const float vmax = red[0];
    __syncthreads();

    float lsum = 0.0f;
    for (int m = tid; m < N_; m += 256) {
        float ev = __expf(row[m] - vmax);
        row[m] = ev;
        lsum += ev;
    }
    red[tid] = lsum; __syncthreads();
    for (int s = 128; s > 0; s >>= 1) {
        if (tid < s) red[tid] += red[tid + s];
        __syncthreads();
    }
    const float inv = 1.0f / red[0];
    __syncthreads();

    for (int m = tid; m < N_; m += 256)
        S[(size_t)n * N_ + m] = (_Float16)(row[m] * inv);
}

// ---------------------------------------------------------------------------
// Kernel 2: node-major pack  INT[m][b*66 + c] = (c<2 ? x[b,m,c] : state[b,m,c-2])
// ---------------------------------------------------------------------------
__global__ __launch_bounds__(256)
void build_in0_kernel(const float* __restrict__ x, const float* __restrict__ st,
                      _Float16* __restrict__ INT_)
{
    int idx = blockIdx.x * 256 + threadIdx.x;
    const int total = N_ * COLS_;
    if (idx >= total) return;
    int m  = idx / COLS_;
    int rc = idx - m * COLS_;
    int b  = rc / CIN_;
    int c  = rc - b * CIN_;
    float v = (c < DIN_) ? x[(b * N_ + m) * DIN_ + c]
                         : st[(b * N_ + m) * DOUT_ + (c - DIN_)];
    INT_[idx] = (_Float16)v;
}

// ---------------------------------------------------------------------------
// Kernel 3: AGG[2048,4224] = S[2048,2048] @ Bsrc[2048,4224], f16 in, f32 acc.
// Block tile 128x128, K-step 32, double-buffered LDS.
// A tile staged asynchronously (global_load_async_to_lds_b128, ASYNCcnt);
// B tile staged through VGPRs because it is transposed into [col][k] layout.
// ---------------------------------------------------------------------------
__global__ __launch_bounds__(256)
void agg_gemm_kernel(const _Float16* __restrict__ S,
                     const _Float16* __restrict__ Bsrc,
                     _Float16* __restrict__ Cdst)
{
    __shared__ _Float16 sA[2][128 * 32];   // row-major [r][k]
    __shared__ _Float16 sB[2][128 * 32];   // col-major [c][k]
    const int tid  = threadIdx.x;
    const int lane = tid & 31;
    const int w    = tid >> 5;
    const int nb   = blockIdx.x * 128;  // column base in [0,4224)
    const int mb   = blockIdx.y * 128;  // row base in [0,2048)
    const int wm   = (w & 3) * 32;      // wave M band (2 subtiles)
    const int wn   = (w >> 2) * 64;     // wave N band (4 subtiles)
    const int lr   = lane & 15;
    const int hi   = lane >> 4;         // 0/1: K-phase select per WMMA layout

    // B-tile staging coordinates (2 chunks of 8 halves per thread)
    const int bk0 = tid >> 4;           // k row (iter 0), +16 for iter 1
    const int bc0 = (tid & 15) * 8;     // first of 8 columns

    v8f acc[2][4];
#pragma unroll
    for (int s = 0; s < 2; ++s)
#pragma unroll
        for (int t = 0; t < 4; ++t) acc[s][t] = {};

    // --- async stage of A tile (128 rows x 32 halves = 512 x 16B chunks) ---
    auto stage_a_async = [&](int buf, int kk) {
        // 2 chunks per thread; chunk ch -> row = ch/4, half-off = (ch%4)*8
#pragma unroll
        for (int q = 0; q < 2; ++q) {
            int ch = tid * 2 + q;
            int r  = ch >> 2;
            int ko = (ch & 3) * 8;
            unsigned long long ga =
                (unsigned long long)(uintptr_t)(S + (size_t)(mb + r) * N_ + kk + ko);
            unsigned lds = (unsigned)(uintptr_t)(&sA[buf][r * 32 + ko]);
            asm volatile("global_load_async_to_lds_b128 %0, %1, off"
                         :: "v"(lds), "v"(ga) : "memory");
        }
    };

    _Float16 breg[2][8];
    auto load_b_regs = [&](int kk) {
#pragma unroll
        for (int it = 0; it < 2; ++it) {
            const _Float16* src =
                Bsrc + (size_t)(kk + it * 16 + bk0) * COLS_ + nb + bc0;
#pragma unroll
            for (int j = 0; j < 8; ++j) breg[it][j] = src[j];
        }
    };

    // prologue
    stage_a_async(0, 0);
    load_b_regs(0);

    const int KSTEPS = N_ / 32;   // 64
    for (int step = 0; step < KSTEPS; ++step) {
        const int cur = step & 1;

        // transpose-store B(cur) from registers into LDS [c][k]
#pragma unroll
        for (int it = 0; it < 2; ++it) {
            int k = it * 16 + bk0;
#pragma unroll
            for (int j = 0; j < 8; ++j) sB[cur][(bc0 + j) * 32 + k] = breg[it][j];
        }
        // our async A(cur) must have landed in LDS before anyone reads it
        asm volatile("s_wait_asynccnt 0x0" ::: "memory");
        __syncthreads();

        // prefetch next K-slice while we compute
        if (step + 1 < KSTEPS) {
            stage_a_async(cur ^ 1, (step + 1) * 32);
            load_b_regs((step + 1) * 32);
        }

        // fragments + WMMA on buffer `cur`
        v16h a[2];
#pragma unroll
        for (int s = 0; s < 2; ++s) {
            const _Float16* p0 = &sA[cur][(wm + s * 16 + lr) * 32 + hi * 8];
            const _Float16* p1 = p0 + 16;
#pragma unroll
            for (int j = 0; j < 8; ++j) { a[s][j] = p0[j]; a[s][8 + j] = p1[j]; }
        }
#pragma unroll
        for (int t = 0; t < 4; ++t) {
            const _Float16* p = &sB[cur][(wn + t * 16 + lr) * 32 + hi * 8];
            v16h bf;
#pragma unroll
            for (int j = 0; j < 8; ++j) { bf[j] = p[j]; bf[8 + j] = p[16 + j]; }
#pragma unroll
            for (int s = 0; s < 2; ++s)
                acc[s][t] = __builtin_amdgcn_wmma_f32_16x16x32_f16(
                    false, a[s], false, bf, (short)0, acc[s][t], false, false);
        }
        __syncthreads();
    }

    // store D: row = base + vgpr + 8*hi, col = base + (lane&15)
#pragma unroll
    for (int s = 0; s < 2; ++s)
#pragma unroll
        for (int t = 0; t < 4; ++t) {
            int col = nb + wn + t * 16 + lr;
#pragma unroll
            for (int v = 0; v < 8; ++v) {
                int r = mb + wm + s * 16 + hi * 8 + v;
                Cdst[(size_t)r * COLS_ + col] = (_Float16)acc[s][t][v];
            }
        }
}

// ---------------------------------------------------------------------------
// Kernel 4 (gate): per node n, Z[64,128] = A_n[64,160] @ W_n[160,128] + bias,
// z_r = sigmoid(Z); writes r (f32) and candidate input concat(x, z*state) to
// IN1 (node-major f16) for the second aggregation GEMM.
// ---------------------------------------------------------------------------
__global__ __launch_bounds__(256)
void gate_node_kernel(const float* __restrict__ x, const float* __restrict__ st,
                      const float* __restrict__ E, const float* __restrict__ wp,
                      const float* __restrict__ bp, const _Float16* __restrict__ AGG,
                      _Float16* __restrict__ IN1, float* __restrict__ R)
{
    __shared__ _Float16 sW[COUTG_ * FP_];  // col-major [o][f], 40KB
    __shared__ _Float16 sA[B_ * FP_];      // row-major [b][f], 20KB
    const int n   = blockIdx.x;
    const int tid = threadIdx.x;
    float e[DD_];
#pragma unroll
    for (int d = 0; d < DD_; ++d) e[d] = E[n * DD_ + d];

    // W_n[f][o] = sum_d e[d] * wp[(d*132 + f)*128 + o]
    for (int t = tid; t < FTOT_ * COUTG_; t += 256) {
        int f = t / COUTG_;
        int o = t - f * COUTG_;
        float acc = 0.0f;
#pragma unroll
        for (int d = 0; d < DD_; ++d) acc += e[d] * wp[(d * FTOT_ + f) * COUTG_ + o];
        sW[o * FP_ + f] = (_Float16)acc;
    }
    for (int t = tid; t < (FP_ - FTOT_) * COUTG_; t += 256) {
        int f = FTOT_ + t / COUTG_;
        int o = t % COUTG_;
        sW[o * FP_ + f] = (_Float16)0.0f;
    }
    // A_n[b][f] = { x | state | agg | 0 }
    for (int t = tid; t < B_ * FP_; t += 256) {
        int b = t / FP_;
        int f = t - b * FP_;
        float v;
        if      (f < DIN_)  v = x[(b * N_ + n) * DIN_ + f];
        else if (f < CIN_)  v = st[(b * N_ + n) * DOUT_ + (f - DIN_)];
        else if (f < FTOT_) v = (float)AGG[(size_t)n * COLS_ + b * CIN_ + (f - CIN_)];
        else                v = 0.0f;
        sA[b * FP_ + f] = (_Float16)v;
    }
    __syncthreads();

    const int lane = tid & 31;
    const int w    = tid >> 5;
    const int lr   = lane & 15;
    const int hi   = lane >> 4;
    const int mi   = (w & 3) * 16;     // row-tile base (batch)
    const int nt0  = (w >> 2) * 4;     // first of 4 N-tiles

    v8f acc[4];
#pragma unroll
    for (int t = 0; t < 4; ++t) acc[t] = {};

#pragma unroll
    for (int ks = 0; ks < FP_; ks += 32) {
        v16h a;
        const _Float16* p0 = sA + (mi + lr) * FP_ + ks + hi * 8;
        const _Float16* p1 = p0 + 16;
#pragma unroll
        for (int j = 0; j < 8; ++j) { a[j] = p0[j]; a[8 + j] = p1[j]; }
#pragma unroll
        for (int t = 0; t < 4; ++t) {
            const _Float16* p = sW + ((nt0 + t) * 16 + lr) * FP_ + ks + hi * 8;
            v16h bf;
#pragma unroll
            for (int j = 0; j < 8; ++j) { bf[j] = p[j]; bf[8 + j] = p[16 + j]; }
            acc[t] = __builtin_amdgcn_wmma_f32_16x16x32_f16(
                false, a, false, bf, (short)0, acc[t], false, false);
        }
    }

    // epilogue: sigmoid(acc + bias); split z (cols<64) / r (cols>=64)
#pragma unroll
    for (int t = 0; t < 4; ++t) {
        int o = (nt0 + t) * 16 + lr;
        float bias = 0.0f;
#pragma unroll
        for (int d = 0; d < DD_; ++d) bias += e[d] * bp[d * COUTG_ + o];
#pragma unroll
        for (int v = 0; v < 8; ++v) {
            int b = mi + hi * 8 + v;
            float val = 1.0f / (1.0f + __expf(-(acc[t][v] + bias)));
            if (o < DOUT_) {  // z: candidate input = z * state
                float s0 = st[(b * N_ + n) * DOUT_ + o];
                IN1[(size_t)n * COLS_ + b * CIN_ + DIN_ + o] = (_Float16)(val * s0);
            } else {          // r: keep f32 for final combine
                R[((size_t)n * B_ + b) * DOUT_ + (o - DOUT_)] = val;
            }
        }
    }
    // x portion of candidate input
    if (tid < B_ * DIN_) {
        int b = tid >> 1, c = tid & 1;
        IN1[(size_t)n * COLS_ + b * CIN_ + c] = (_Float16)x[(b * N_ + n) * DIN_ + c];
    }
}

// ---------------------------------------------------------------------------
// Kernel 5 (update): hc = tanh(A_n[64,160] @ Wu_n[160,64] + bias);
// h = r*state + (1-r)*hc -> d_out (f32, [B,N,64]).
// ---------------------------------------------------------------------------
__global__ __launch_bounds__(256)
void update_node_kernel(const float* __restrict__ st, const float* __restrict__ E,
                        const float* __restrict__ wp, const float* __restrict__ bp,
                        const _Float16* __restrict__ IN1, const _Float16* __restrict__ AGG,
                        const float* __restrict__ R, float* __restrict__ out)
{
    __shared__ _Float16 sW[COUTU_ * FP_];  // 20KB
    __shared__ _Float16 sA[B_ * FP_];      // 20KB
    const int n   = blockIdx.x;
    const int tid = threadIdx.x;
    float e[DD_];
#pragma unroll
    for (int d = 0; d < DD_; ++d) e[d] = E[n * DD_ + d];

    for (int t = tid; t < FTOT_ * COUTU_; t += 256) {
        int f = t / COUTU_;
        int o = t - f * COUTU_;
        float acc = 0.0f;
#pragma unroll
        for (int d = 0; d < DD_; ++d) acc += e[d] * wp[(d * FTOT_ + f) * COUTU_ + o];
        sW[o * FP_ + f] = (_Float16)acc;
    }
    for (int t = tid; t < (FP_ - FTOT_) * COUTU_; t += 256) {
        int f = FTOT_ + t / COUTU_;
        int o = t % COUTU_;
        sW[o * FP_ + f] = (_Float16)0.0f;
    }
    for (int t = tid; t < B_ * FP_; t += 256) {
        int b = t / FP_;
        int f = t - b * FP_;
        _Float16 v;
        if      (f < CIN_)  v = IN1[(size_t)n * COLS_ + b * CIN_ + f];
        else if (f < FTOT_) v = AGG[(size_t)n * COLS_ + b * CIN_ + (f - CIN_)];
        else                v = (_Float16)0.0f;
        sA[b * FP_ + f] = v;
    }
    __syncthreads();

    const int lane = tid & 31;
    const int w    = tid >> 5;
    const int lr   = lane & 15;
    const int hi   = lane >> 4;
    const int mi   = (w & 3) * 16;
    const int nt0  = (w >> 2) * 2;   // 2 N-tiles per wave (4x4 tiles, 8 waves)

    v8f acc[2];
#pragma unroll
    for (int t = 0; t < 2; ++t) acc[t] = {};

#pragma unroll
    for (int ks = 0; ks < FP_; ks += 32) {
        v16h a;
        const _Float16* p0 = sA + (mi + lr) * FP_ + ks + hi * 8;
        const _Float16* p1 = p0 + 16;
#pragma unroll
        for (int j = 0; j < 8; ++j) { a[j] = p0[j]; a[8 + j] = p1[j]; }
#pragma unroll
        for (int t = 0; t < 2; ++t) {
            const _Float16* p = sW + ((nt0 + t) * 16 + lr) * FP_ + ks + hi * 8;
            v16h bf;
#pragma unroll
            for (int j = 0; j < 8; ++j) { bf[j] = p[j]; bf[8 + j] = p[16 + j]; }
            acc[t] = __builtin_amdgcn_wmma_f32_16x16x32_f16(
                false, a, false, bf, (short)0, acc[t], false, false);
        }
    }

#pragma unroll
    for (int t = 0; t < 2; ++t) {
        int o = (nt0 + t) * 16 + lr;
        float bias = 0.0f;
#pragma unroll
        for (int d = 0; d < DD_; ++d) bias += e[d] * bp[d * COUTU_ + o];
#pragma unroll
        for (int v = 0; v < 8; ++v) {
            int b = mi + hi * 8 + v;
            float hc = tanhf(acc[t][v] + bias);
            float r  = R[((size_t)n * B_ + b) * DOUT_ + o];
            float s0 = st[(b * N_ + n) * DOUT_ + o];
            out[((size_t)b * N_ + n) * DOUT_ + o] = r * s0 + (1.0f - r) * hc;
        }
    }
}

// ---------------------------------------------------------------------------
// Launch. Workspace layout (bytes):
//   S    f16 [2048][2048]      :  8,388,608
//   INT  f16 [2048][4224]      : 17,301,504  (in0^T, then reused as cand^T)
//   AGG  f16 [2048][4224]      : 17,301,504
//   R    f32 [2048][64][64]    : 33,554,432
// total ~76.6 MB
// ---------------------------------------------------------------------------
extern "C" void kernel_launch(void* const* d_in, const int* in_sizes, int n_in,
                              void* d_out, int out_size, void* d_ws, size_t ws_size,
                              hipStream_t stream)
{
    (void)in_sizes; (void)n_in; (void)out_size; (void)ws_size;
    const float* x   = (const float*)d_in[0];
    const float* st  = (const float*)d_in[1];
    const float* E   = (const float*)d_in[2];
    const float* gwp = (const float*)d_in[3];
    const float* gbp = (const float*)d_in[4];
    const float* uwp = (const float*)d_in[5];
    const float* ubp = (const float*)d_in[6];
    float* out = (float*)d_out;

    char* ws = (char*)d_ws;
    _Float16* S    = (_Float16*)(ws);
    _Float16* INT_ = (_Float16*)(ws + 8388608);
    _Float16* AGG  = (_Float16*)(ws + 8388608 + 17301504);
    float*    R    = (float*)   (ws + 8388608 + 2 * 17301504);

    supports_softmax_kernel<<<N_, 256, 0, stream>>>(E, S);
    build_in0_kernel<<<(N_ * COLS_ + 255) / 256, 256, 0, stream>>>(x, st, INT_);
    agg_gemm_kernel<<<dim3(COLS_ / 128, N_ / 128), 256, 0, stream>>>(S, INT_, AGG);
    gate_node_kernel<<<N_, 256, 0, stream>>>(x, st, E, gwp, gbp, AGG, INT_, R);
    agg_gemm_kernel<<<dim3(COLS_ / 128, N_ / 128), 256, 0, stream>>>(S, INT_, AGG);
    update_node_kernel<<<N_, 256, 0, stream>>>(st, E, uwp, ubp, INT_, AGG, R, out);
}
